// MLPHeb_36704790511730
// MI455X (gfx1250) — 16-bit and 8-bit matrix operand order repaired and checked by reading, compiled, not run on hardware
//
#include <hip/hip_runtime.h>
#include <hip/hip_bf16.h>


typedef __bf16 v16bf __attribute__((ext_vector_type(16)));
typedef __bf16 v8bf  __attribute__((ext_vector_type(8)));
typedef float  v8f   __attribute__((ext_vector_type(8)));
typedef float  v4f   __attribute__((ext_vector_type(4)));
typedef int    v8i   __attribute__((ext_vector_type(8)));
typedef int    v4i   __attribute__((ext_vector_type(4)));
typedef int    v2i   __attribute__((ext_vector_type(2)));

#define BATCH 32768
#define DIN   784
#define DINP  800
#define HID   256
#define NCLS  10

#define WMMA_BF16(a, b, c) \
    __builtin_amdgcn_wmma_f32_16x16x32_bf16(false, (a), false, (b), (short)0, (c), false, false)
#define WMMA_IU8(a, b, c) \
    __builtin_amdgcn_wmma_i32_16x16x64_iu8(false, (a), false, (b), (c), false, false)

__device__ inline v16bf loadA_bf16(const __bf16* base, int ld, int m, int k0) {
    const int lane = threadIdx.x & 31;
    const __bf16* p = base + (size_t)(m + (lane & 15)) * ld + k0 + ((lane & 16) ? 8 : 0);
    v8bf lo = *(const v8bf*)p;
    v8bf hi = *(const v8bf*)(p + 16);
    v16bf r;
#pragma unroll
    for (int i = 0; i < 8; ++i) { r[i] = lo[i]; r[i + 8] = hi[i]; }
    return r;
}

__device__ inline v16bf loadA_f32(const float* X, int m, int k0) {
    const int lane = threadIdx.x & 31;
    const float* p = X + (size_t)(m + (lane & 15)) * DIN;
    const int ka = k0 + ((lane & 16) ? 8 : 0);
    const int kb = ka + 16;
    v16bf r;
    if (ka < DIN) {
        v4f a = *(const v4f*)(p + ka);
        v4f b = *(const v4f*)(p + ka + 4);
#pragma unroll
        for (int i = 0; i < 4; ++i) { r[i] = (__bf16)a[i]; r[4 + i] = (__bf16)b[i]; }
    } else {
#pragma unroll
        for (int i = 0; i < 8; ++i) r[i] = (__bf16)0.f;
    }
    if (kb < DIN) {
        v4f a = *(const v4f*)(p + kb);
        v4f b = *(const v4f*)(p + kb + 4);
#pragma unroll
        for (int i = 0; i < 4; ++i) { r[8 + i] = (__bf16)a[i]; r[12 + i] = (__bf16)b[i]; }
    } else {
#pragma unroll
        for (int i = 0; i < 8; ++i) r[8 + i] = (__bf16)0.f;
    }
    return r;
}

__device__ inline v16bf loadB_bf16(const __bf16* base, int ld, int n, int k0) {
    const int lane = threadIdx.x & 31;
    const __bf16* p = base + (size_t)(n + (lane & 15)) * ld + k0 + ((lane & 16) ? 8 : 0);
    v8bf lo = *(const v8bf*)p;
    v8bf hi = *(const v8bf*)(p + 16);
    v16bf r;
#pragma unroll
    for (int i = 0; i < 8; ++i) { r[i] = lo[i]; r[i + 8] = hi[i]; }
    return r;
}

__device__ inline v8i loadA_u8(const unsigned char* base, int m, int k0) {
    const int lane = threadIdx.x & 31;
    const unsigned char* p = base + (size_t)(m + (lane & 15)) * BATCH + k0 + ((lane & 16) ? 8 : 0);
    v8i r; v2i t;
    t = *(const v2i*)(p);      r[0] = t[0]; r[1] = t[1];
    t = *(const v2i*)(p + 16); r[2] = t[0]; r[3] = t[1];
    t = *(const v2i*)(p + 32); r[4] = t[0]; r[5] = t[1];
    t = *(const v2i*)(p + 48); r[6] = t[0]; r[7] = t[1];
    return r;
}

__device__ inline v8i loadB_u8(const unsigned char* base, int n, int k0) {
    const int lane = threadIdx.x & 31;
    const unsigned char* p = base + (size_t)(n + (lane & 15)) * BATCH + k0 + ((lane & 16) ? 8 : 0);
    v8i r; v2i t;
    t = *(const v2i*)(p);      r[0] = t[0]; r[1] = t[1];
    t = *(const v2i*)(p + 16); r[2] = t[0]; r[3] = t[1];
    t = *(const v2i*)(p + 32); r[4] = t[0]; r[5] = t[1];
    t = *(const v2i*)(p + 48); r[6] = t[0]; r[7] = t[1];
    return r;
}

__device__ inline void epilogue_tile(v8f c, int m0, int n0, const float* bias,
                                     __bf16* Hout, unsigned char* aT) {
    const int lane = threadIdx.x & 31;
    const int n  = n0 + (lane & 15);
    const int mB = m0 + ((lane & 16) ? 8 : 0);
    const float bn = bias[n];
    unsigned long long pk = 0ull;
#pragma unroll
    for (int i = 0; i < 8; ++i) {
        float v = c[i] + bn;
        v = v > 0.f ? v : 0.f;
        Hout[(size_t)(mB + i) * HID + n] = (__bf16)v;
        pk |= (unsigned long long)(v > 0.f ? 1u : 0u) << (8 * i);
    }
    *(unsigned long long*)(aT + (size_t)n * BATCH + mB) = pk;
}

__device__ inline void coact_commit(v8i acc, int m0, int n0, float* out,
                                    int Mvalid, int Nvalid, int ldo) {
    const int lane = threadIdx.x & 31;
    const int n  = n0 + (lane & 15);
    const int mB = m0 + ((lane & 16) ? 8 : 0);
    if (n < Nvalid) {
#pragma unroll
        for (int i = 0; i < 8; ++i) {
            const int m = mB + i;
            if (m < Mvalid)
                __hip_atomic_fetch_add(&out[(size_t)m * ldo + n], (float)acc[i],
                                       __ATOMIC_RELAXED, __HIP_MEMORY_SCOPE_AGENT);
        }
    }
}


__global__ void zero_f32(float* p, int n) {
    int i = blockIdx.x * 256 + threadIdx.x;
    if (i < n) p[i] = 0.f;
}

__global__ void prep_w(const float* __restrict__ W1, const float* __restrict__ W2,
                       const float* __restrict__ W3, const float* __restrict__ W4,
                       __bf16* __restrict__ W1b, __bf16* __restrict__ W2b,
                       __bf16* __restrict__ W3b, __bf16* __restrict__ W4b) {
    int i = blockIdx.x * 256 + threadIdx.x;
    if (i < HID * DINP) {
        int r = i / DINP, k = i % DINP;
        W1b[i] = (k < DIN) ? (__bf16)W1[(size_t)r * DIN + k] : (__bf16)0.f;
        return;
    }
    i -= HID * DINP;
    if (i < HID * HID) { W2b[i] = (__bf16)W2[i]; return; }
    i -= HID * HID;
    if (i < HID * HID) { W3b[i] = (__bf16)W3[i]; return; }
    i -= HID * HID;
    if (i < 16 * HID) {
        int r = i / HID;
        W4b[i] = (r < NCLS) ? (__bf16)W4[i] : (__bf16)0.f;
    }
}

__global__ __launch_bounds__(256) void make_a0T(const float* __restrict__ x,
                                                unsigned char* __restrict__ a0T) {
    __shared__ unsigned char t[16][17];
    const int b0 = blockIdx.x * 16, f0 = blockIdx.y * 16;
    const int fx = threadIdx.x & 15, by = threadIdx.x >> 4;
    t[by][fx] = (x[(size_t)(b0 + by) * DIN + f0 + fx] > 0.f) ? 1 : 0;
    __syncthreads();
    const int bx = threadIdx.x & 15, fy = threadIdx.x >> 4;
    a0T[(size_t)(f0 + fy) * BATCH + b0 + bx] = t[bx][fy];
}

__global__ __launch_bounds__(256) void layer1_kernel(
        const float* __restrict__ x, const __bf16* __restrict__ W,
        const float* __restrict__ bias, __bf16* __restrict__ Hout,
        unsigned char* __restrict__ aT) {
    const int wave = threadIdx.x >> 5;
    const int lane = threadIdx.x & 31;
    const int m0 = blockIdx.x * 128 + (wave >> 1) * 32;
    const int n0 = blockIdx.y * 64 + (wave & 1) * 32;
    v8f acc00 = {}, acc01 = {}, acc10 = {}, acc11 = {};
    for (int k = 0; k < DINP; k += 32) {
        if (k + 32 < DIN)
            __builtin_prefetch(x + (size_t)(m0 + (lane & 15)) * DIN + k + 32, 0, 1);
        v16bf a0 = loadA_f32(x, m0, k);
        v16bf a1 = loadA_f32(x, m0 + 16, k);
        v16bf b0 = loadB_bf16(W, DINP, n0, k);
        v16bf b1 = loadB_bf16(W, DINP, n0 + 16, k);
        acc00 = WMMA_BF16(a0, b0, acc00);
        acc01 = WMMA_BF16(a0, b1, acc01);
        acc10 = WMMA_BF16(a1, b0, acc10);
        acc11 = WMMA_BF16(a1, b1, acc11);
    }
    epilogue_tile(acc00, m0,      n0,      bias, Hout, aT);
    epilogue_tile(acc01, m0,      n0 + 16, bias, Hout, aT);
    epilogue_tile(acc10, m0 + 16, n0,      bias, Hout, aT);
    epilogue_tile(acc11, m0 + 16, n0 + 16, bias, Hout, aT);
}

__global__ __launch_bounds__(256) void layer_mid_kernel(
        const __bf16* __restrict__ Hin, const __bf16* __restrict__ W,
        const float* __restrict__ bias, __bf16* __restrict__ Hout,
        unsigned char* __restrict__ aT) {
    const int wave = threadIdx.x >> 5;
    const int m0 = blockIdx.x * 128 + (wave >> 1) * 32;
    const int n0 = blockIdx.y * 64 + (wave & 1) * 32;
    v8f acc00 = {}, acc01 = {}, acc10 = {}, acc11 = {};
    for (int k = 0; k < HID; k += 32) {
        v16bf a0 = loadA_bf16(Hin, HID, m0, k);
        v16bf a1 = loadA_bf16(Hin, HID, m0 + 16, k);
        v16bf b0 = loadB_bf16(W, HID, n0, k);
        v16bf b1 = loadB_bf16(W, HID, n0 + 16, k);
        acc00 = WMMA_BF16(a0, b0, acc00);
        acc01 = WMMA_BF16(a0, b1, acc01);
        acc10 = WMMA_BF16(a1, b0, acc10);
        acc11 = WMMA_BF16(a1, b1, acc11);
    }
    epilogue_tile(acc00, m0,      n0,      bias, Hout, aT);
    epilogue_tile(acc01, m0,      n0 + 16, bias, Hout, aT);
    epilogue_tile(acc10, m0 + 16, n0,      bias, Hout, aT);
    epilogue_tile(acc11, m0 + 16, n0 + 16, bias, Hout, aT);
}

__global__ __launch_bounds__(256) void layer4_kernel(
        const __bf16* __restrict__ H3, const __bf16* __restrict__ W4b,
        const float* __restrict__ b4, float* __restrict__ logits,
        unsigned char* __restrict__ a4T) {
    const int wave = threadIdx.x >> 5;
    const int lane = threadIdx.x & 31;
    const int m0 = blockIdx.x * 128 + wave * 16;
    v8f acc = {};
    for (int k = 0; k < HID; k += 32) {
        v16bf a = loadA_bf16(H3, HID, m0, k);
        v16bf b = loadB_bf16(W4b, HID, 0, k);
        acc = WMMA_BF16(a, b, acc);
    }
    const int n  = lane & 15;
    const int mB = m0 + ((lane & 16) ? 8 : 0);
    const float bn = (n < NCLS) ? b4[n] : 0.f;
    unsigned long long pk = 0ull;
#pragma unroll
    for (int i = 0; i < 8; ++i) {
        float v = acc[i] + bn;
        if (n < NCLS) logits[(size_t)(mB + i) * NCLS + n] = v;
        pk |= (unsigned long long)(v > 0.f ? 1u : 0u) << (8 * i);
    }
    *(unsigned long long*)(a4T + (size_t)n * BATCH + mB) = pk;
}

__global__ __launch_bounds__(256) void coact32_kernel(
        const unsigned char* __restrict__ aPrevT, const unsigned char* __restrict__ aCurT,
        float* __restrict__ out, int MT2, int NT2, int Mvalid, int Nvalid, int ldo) {
    const int wave = threadIdx.x >> 5;
    const int lane = threadIdx.x & 31;
    const int task = blockIdx.x * 8 + wave;
    const int chunk = task & 31;
    const int tile  = task >> 5;
    if (tile >= MT2 * NT2) return;
    const int m0 = (tile / NT2) * 32;
    const int n0 = (tile % NT2) * 32;
    const int kb = chunk * (BATCH / 32);
    v8i acc00 = {}, acc01 = {}, acc10 = {}, acc11 = {};
#pragma unroll 2
    for (int k = 0; k < BATCH / 32; k += 64) {
        __builtin_prefetch(aPrevT + (size_t)(m0 + (lane & 15)) * BATCH + kb + k + 64, 0, 1);
        __builtin_prefetch(aCurT + (size_t)(n0 + (lane & 15)) * BATCH + kb + k + 64, 0, 1);
        v8i A0 = loadA_u8(aPrevT, m0,      kb + k);
        v8i A1 = loadA_u8(aPrevT, m0 + 16, kb + k);
        v8i B0 = loadB_u8(aCurT,  n0,      kb + k);
        v8i B1 = loadB_u8(aCurT,  n0 + 16, kb + k);
        acc00 = WMMA_IU8(A0, B0, acc00);
        acc01 = WMMA_IU8(A0, B1, acc01);
        acc10 = WMMA_IU8(A1, B0, acc10);
        acc11 = WMMA_IU8(A1, B1, acc11);
    }
    coact_commit(acc00, m0,      n0,      out, Mvalid, Nvalid, ldo);
    coact_commit(acc01, m0,      n0 + 16, out, Mvalid, Nvalid, ldo);
    coact_commit(acc10, m0 + 16, n0,      out, Mvalid, Nvalid, ldo);
    coact_commit(acc11, m0 + 16, n0 + 16, out, Mvalid, Nvalid, ldo);
}

__global__ __launch_bounds__(256) void coact16_kernel(
        const unsigned char* __restrict__ aPrevT, const unsigned char* __restrict__ aCurT,
        float* __restrict__ out, int MT, int NT, int Mvalid, int Nvalid, int ldo) {
    const int wave = threadIdx.x >> 5;
    const int task = blockIdx.x * 8 + wave;
    const int chunk = task & 31;
    const int tile  = task >> 5;
    if (tile >= MT * NT) return;
    const int m0 = (tile / NT) * 16;
    const int n0 = (tile % NT) * 16;
    const int kb = chunk * (BATCH / 32);
    v8i acc = {};
#pragma unroll 4
    for (int k = 0; k < BATCH / 32; k += 64) {
        v8i A  = loadA_u8(aPrevT, m0, kb + k);
        v8i Bv = loadB_u8(aCurT, n0, kb + k);
        acc = WMMA_IU8(A, Bv, acc);
    }
    coact_commit(acc, m0, n0, out, Mvalid, Nvalid, ldo);
}


extern "C" void kernel_launch(void* const* d_in, const int* in_sizes, int n_in,
                              void* d_out, int out_size, void* d_ws, size_t ws_size,
                              hipStream_t stream) {
    (void)in_sizes; (void)n_in; (void)out_size; (void)ws_size;
    const float* x  = (const float*)d_in[0];
    const float* W1 = (const float*)d_in[1];
    const float* b1 = (const float*)d_in[2];
    const float* W2 = (const float*)d_in[3];
    const float* b2 = (const float*)d_in[4];
    const float* W3 = (const float*)d_in[5];
    const float* b3 = (const float*)d_in[6];
    const float* W4 = (const float*)d_in[7];
    const float* b4 = (const float*)d_in[8];
    float* out = (float*)d_out;
    char*  ws  = (char*)d_ws;

    __bf16* W1b = (__bf16*)(ws + 0);
    __bf16* W2b = (__bf16*)(ws + 409600);
    __bf16* W3b = (__bf16*)(ws + 540672);
    __bf16* W4b = (__bf16*)(ws + 671744);
    __bf16* h1b = (__bf16*)(ws + 679936);
    __bf16* h2b = (__bf16*)(ws + 17457152);
    unsigned char* a0T = (unsigned char*)(ws + 34234368);
    unsigned char* a1T = (unsigned char*)(ws + 59924480);
    unsigned char* a2T = (unsigned char*)(ws + 68313088);
    unsigned char* a3T = (unsigned char*)(ws + 76701696);
    unsigned char* a4T = (unsigned char*)(ws + 85090304);
    __bf16* h3b = h1b;

    float* logits = out;
    float* c0 = out + 327680;
    float* c1 = out + 528384;
    float* c2 = out + 593920;
    float* c3 = out + 659456;

    zero_f32<<<2586, 256, 0, stream>>>(out, 662016);
    prep_w<<<1328, 256, 0, stream>>>(W1, W2, W3, W4, W1b, W2b, W3b, W4b);
    make_a0T<<<dim3(BATCH / 16, DIN / 16), 256, 0, stream>>>(x, a0T);

    layer1_kernel<<<dim3(BATCH / 128, HID / 64), 256, 0, stream>>>(x, W1b, b1, h1b, a1T);
    layer_mid_kernel<<<dim3(BATCH / 128, HID / 64), 256, 0, stream>>>(h1b, W2b, b2, h2b, a2T);
    layer_mid_kernel<<<dim3(BATCH / 128, HID / 64), 256, 0, stream>>>(h2b, W3b, b3, h3b, a3T);
    layer4_kernel<<<BATCH / 128, 256, 0, stream>>>(h3b, W4b, b4, logits, a4T);

    coact32_kernel<<<(25 * 8 * 32) / 8, 256, 0, stream>>>(a0T, a1T, c0, 25, 8, DIN, HID, HID);
    coact32_kernel<<<(8 * 8 * 32) / 8, 256, 0, stream>>>(a1T, a2T, c1, 8, 8, HID, HID, HID);
    coact32_kernel<<<(8 * 8 * 32) / 8, 256, 0, stream>>>(a2T, a3T, c2, 8, 8, HID, HID, HID);
    coact16_kernel<<<(16 * 1 * 32) / 8, 256, 0, stream>>>(a3T, a4T, c3, 16, 1, HID, NCLS, NCLS);
}
